// BaseLineModel_75067438399858
// MI455X (gfx1250) — compile-verified, matrix operand order checked by reading, not run on hardware
//
#include <hip/hip_runtime.h>
#include <hip/hip_bf16.h>
#include <math.h>

// ---------------------------------------------------------------------------
// Problem constants (from reference)
// ---------------------------------------------------------------------------
#define B_      32
#define S_      64
#define L_      128
#define TOTAL_  1024
#define E_      300
#define F_      256
#define W_      5
#define H_      256
#define T_      124              // L - W + 1
#define TP_     128              // T padded to multiple of 16
#define KCONV_  1500             // W * E
#define KPCONV_ 1504             // K padded to multiple of 32
#define SSTRIDE_ 39904           // L*E + KPCONV (flattened emb + window slack)

#define LDSROW_ 40               // padded LDS row stride (elements) per column
#define LDSBUF_ (256 * LDSROW_)  // 10240 elements per buffer

typedef __bf16 bf16;
typedef __attribute__((ext_vector_type(16))) __bf16 bf16x16;
typedef __attribute__((ext_vector_type(4)))  __bf16 bf16x4;
typedef __attribute__((ext_vector_type(8)))  float  f32x8;

__device__ __forceinline__ float sigf(float x) { return 1.0f / (1.0f + expf(-x)); }

// Load 8 contiguous bf16 (>=8B-aligned) into fragment elements [off, off+8)
__device__ __forceinline__ void load8(bf16x16& f, int off, const bf16* p) {
    const bf16x4* q = reinterpret_cast<const bf16x4*>(p);
    bf16x4 a = q[0];
    bf16x4 b = q[1];
#pragma unroll
    for (int i = 0; i < 4; ++i) { f[off + i] = a[i]; f[off + 4 + i] = b[i]; }
}

// 16-element bf16 fragment: runs of 8 at p and p+16 (K layout for 16x32 tiles:
// lanes<16 hold K {0..7,16..23}, lanes>=16 hold K {8..15,24..31}; caller folds
// the hi*8 lane offset into p).
__device__ __forceinline__ bf16x16 load_frag(const bf16* p) {
    bf16x16 f;
    load8(f, 0, p);
    load8(f, 8, p + 16);
    return f;
}

// ---------------------------------------------------------------------------
// LDS-staged, double-buffered GEMM worker.
// Block = 512 threads = 16 waves. Wave w: M-tile (w&7), N-tiles (w>>3)*8..+7.
// Per K-chunk (32 wide): the whole block stages B[256 cols][32 k] (16KB) into
// LDS; global loads for chunk kc+1 are issued BEFORE the 8 WMMAs of chunk kc
// so L2 latency hides behind matrix work. B fragments are read from LDS with
// a one-ahead rotation: tile j+1's ds_loads are in flight while the WMMA for
// tile j executes (partial s_wait_dscnt, and no WMMA->VALU WAR-hazard NOPs
// since post-WMMA instructions write the other buffer). LDS rows padded to 40
// elements so the 16-lane 16B reads land in distinct banks.
// ---------------------------------------------------------------------------
template <int KP, int KCHUNKS>
__device__ __forceinline__ void tile_gemm_lds(const bf16* __restrict__ Arow,
                                              const bf16* __restrict__ Bt,
                                              unsigned short (*__restrict__ smB)[LDSBUF_],
                                              int tid, f32x8 acc[8]) {
    const int lane = tid & 31;
    const int hi = lane >> 4, ln = lane & 15;
    const int nhalf = (tid >> 5) >> 3;          // 0 or 1
    const int h8 = hi * 8;
    const int fbase = ln * LDSROW_ + h8;        // in-tile fragment offset
    const int tbase = nhalf * 8 * 16 * LDSROW_; // first tile column offset

    // Staging role: thread -> (column, k-half)
    const int col  = tid & 255;
    const int half = (tid >> 8) & 1;            // 0: k 0..15, 1: k 16..31
    const bf16* gB = Bt + (size_t)col * KP + half * 16;
    const int ldst = col * LDSROW_ + half * 16; // LDS element offset for staging

    // Stage chunk 0 into buffer 0
    {
        uint4 g0 = *(const uint4*)(gB + 0);
        uint4 g1 = *(const uint4*)(gB + 8);
        uint4* d = (uint4*)&smB[0][ldst];
        d[0] = g0; d[1] = g1;
    }
    __syncthreads();

    bf16x16 a = load_frag(Arow + h8);           // A fragment, chunk 0

#pragma unroll 1
    for (int kc = 0; kc < KCHUNKS; ++kc) {
        const int k1  = (kc + 1 < KCHUNKS) ? (kc + 1) * 32 : 0;  // clamped
        const int cur = kc & 1, nxt = cur ^ 1;

        // Issue next-chunk global loads early (latency overlapped with WMMAs)
        uint4 g0 = *(const uint4*)(gB + k1);
        uint4 g1 = *(const uint4*)(gB + k1 + 8);
        __builtin_prefetch(Arow + k1 + 256, 0, 0);               // global_prefetch_b8
        bf16x16 a_next = load_frag(Arow + k1 + h8);

        // Compute 8 N-tiles from current LDS buffer, B-fragment one ahead
        const bf16* lb = (const bf16*)&smB[cur][0];
        bf16x16 bnx = load_frag(lb + tbase + fbase);
#pragma unroll
        for (int j = 0; j < 8; ++j) {
            bf16x16 bc = bnx;                    // loaded one iteration ago
            if (j < 7)
                bnx = load_frag(lb + tbase + (j + 1) * 16 * LDSROW_ + fbase);
            acc[j] = __builtin_amdgcn_wmma_f32_16x16x32_bf16(
                false, a, false, bc, (short)0, acc[j], false, false);
        }

        __syncthreads();                        // all reads of `nxt` (prev iter) done
        {
            uint4* d = (uint4*)&smB[nxt][ldst];
            d[0] = g0; d[1] = g1;
        }
        __syncthreads();                        // staging visible before next reads
        a = a_next;
    }
}

// ---------------------------------------------------------------------------
// Prep kernels
// ---------------------------------------------------------------------------
// Gather embeddings -> flattened bf16 per sentence (zero slack region)
__global__ void k_emb(const int* __restrict__ input, const float* __restrict__ Wemb,
                      bf16* __restrict__ embBf) {
    size_t idx = (size_t)blockIdx.x * 256 + threadIdx.x;
    if (idx >= (size_t)TOTAL_ * SSTRIDE_) return;
    int sent = (int)(idx / SSTRIDE_);
    int off  = (int)(idx % SSTRIDE_);
    float v = 0.0f;
    if (off < L_ * E_) {
        int tok = off / E_, e = off % E_;
        int w = input[sent * L_ + tok];
        v = Wemb[(size_t)w * E_ + e];
    }
    embBf[idx] = (bf16)v;
}

// Wconv [F,1,W,E] -> Bt [F][KPCONV] bf16 (zero K-pad)
__global__ void k_btconv(const float* __restrict__ Wconv, bf16* __restrict__ Bt) {
    int idx = blockIdx.x * 256 + threadIdx.x;
    if (idx >= F_ * KPCONV_) return;
    int f = idx / KPCONV_, k = idx % KPCONV_;
    float v = (k < KCONV_) ? Wconv[(size_t)f * KCONV_ + k] : 0.0f;
    Bt[idx] = (bf16)v;
}

// Wa0 [F][F] -> Bt [n][k] = Wa0[k][n]
__global__ void k_btwa0(const float* __restrict__ Wa0, bf16* __restrict__ Bt) {
    int idx = blockIdx.x * 256 + threadIdx.x;
    if (idx >= F_ * F_) return;
    int n = idx / F_, k = idx % F_;
    Bt[idx] = (bf16)Wa0[k * F_ + n];
}

// Transpose [1024][256] f32 -> [256][1024] f32 (for coalesced LSTM reads)
__global__ void k_transpose_w(const float* __restrict__ src, float* __restrict__ dst) {
    int idx = blockIdx.x * 256 + threadIdx.x;           // idx = j*256 + k
    if (idx >= 1024 * 256) return;
    int j = idx / 256, k = idx % 256;
    dst[k * 1024 + j] = src[idx];
}

// ---------------------------------------------------------------------------
// Phase B: conv-as-GEMM  c0 = tanh(A[128x1504] @ B[1504x256] + bconv)
// A[t][k] = embFlat[t*E + k]  (sliding-window trick, no im2col materialized)
// ---------------------------------------------------------------------------
__global__ __launch_bounds__(512) void k_conv(const bf16* __restrict__ embBf,
                                              const bf16* __restrict__ BtConv,
                                              const float* __restrict__ bconv,
                                              bf16* __restrict__ c0) {
    __shared__ __align__(16) unsigned short smB[2][LDSBUF_];
    int sent = blockIdx.x;
    int tid = threadIdx.x;
    int wave = tid >> 5, lane = tid & 31;
    int mt = wave & 7, nhalf = wave >> 3;
    int hi = lane >> 4, ln = lane & 15;
    const bf16* Arow = embBf + (size_t)sent * SSTRIDE_ + (size_t)(mt * 16 + ln) * E_;
    f32x8 acc[8];
#pragma unroll
    for (int j = 0; j < 8; ++j)
#pragma unroll
        for (int r = 0; r < 8; ++r) acc[j][r] = 0.0f;
    tile_gemm_lds<KPCONV_, KPCONV_ / 32>(Arow, BtConv, smB, tid, acc);
#pragma unroll
    for (int j = 0; j < 8; ++j) {
        int f = (nhalf * 8 + j) * 16 + ln;
        float bb = bconv[f];
#pragma unroll
        for (int r = 0; r < 8; ++r) {
            int t = mt * 16 + hi * 8 + r;               // C/D layout: m = r + 8*hi
            c0[((size_t)sent * TP_ + t) * F_ + f] = (bf16)tanhf(acc[j][r] + bb);
        }
    }
}

// ---------------------------------------------------------------------------
// Phase C: attention-0 GEMM  logits = tanh(c0[128x256] @ Wa0 + ba0)
// ---------------------------------------------------------------------------
__global__ __launch_bounds__(512) void k_attn0(const bf16* __restrict__ c0,
                                               const bf16* __restrict__ BtWa0,
                                               const float* __restrict__ ba0,
                                               bf16* __restrict__ logits) {
    __shared__ __align__(16) unsigned short smB[2][LDSBUF_];
    int sent = blockIdx.x;
    int tid = threadIdx.x;
    int wave = tid >> 5, lane = tid & 31;
    int mt = wave & 7, nhalf = wave >> 3;
    int hi = lane >> 4, ln = lane & 15;
    const bf16* Arow = c0 + ((size_t)sent * TP_ + (mt * 16 + ln)) * F_;
    f32x8 acc[8];
#pragma unroll
    for (int j = 0; j < 8; ++j)
#pragma unroll
        for (int r = 0; r < 8; ++r) acc[j][r] = 0.0f;
    tile_gemm_lds<F_, F_ / 32>(Arow, BtWa0, smB, tid, acc);
#pragma unroll
    for (int j = 0; j < 8; ++j) {
        int f = (nhalf * 8 + j) * 16 + ln;
        float bb = ba0[f];
#pragma unroll
        for (int r = 0; r < 8; ++r) {
            int t = mt * 16 + hi * 8 + r;
            logits[((size_t)sent * TP_ + t) * F_ + f] = (bf16)tanhf(acc[j][r] + bb);
        }
    }
}

// ---------------------------------------------------------------------------
// Phase D: per-feature softmax over t (124 valid rows) + weighted sum -> s0
// ---------------------------------------------------------------------------
__global__ __launch_bounds__(256) void k_soft0(const bf16* __restrict__ logits,
                                               const bf16* __restrict__ c0,
                                               float* __restrict__ s0) {
    int sent = blockIdx.x, f = threadIdx.x;
    float mx = -3.4e38f;
    for (int t = 0; t < T_; ++t) {
        float l = (float)logits[((size_t)sent * TP_ + t) * F_ + f];
        mx = fmaxf(mx, l);
    }
    float d = 0.0f, num = 0.0f;
    for (int t = 0; t < T_; ++t) {
        size_t ix = ((size_t)sent * TP_ + t) * F_ + f;
        float e = expf((float)logits[ix] - mx);
        d += e;
        num += e * (float)c0[ix];
    }
    s0[(size_t)sent * F_ + f] = num / d;
}

// ---------------------------------------------------------------------------
// Phase E: zero t0 then ragged scatter s0 -> t0[B,S,F]
// ---------------------------------------------------------------------------
__global__ void k_zero_t0(float* __restrict__ t0) {
    int idx = blockIdx.x * 256 + threadIdx.x;
    if (idx < B_ * S_ * F_) t0[idx] = 0.0f;
}

__global__ void k_scatter(const float* __restrict__ s0, const int* __restrict__ num_sent,
                          float* __restrict__ t0) {
    int idx = blockIdx.x * 256 + threadIdx.x;          // TOTAL * F
    if (idx >= TOTAL_ * F_) return;
    int i = idx / F_, f = idx % F_;
    int cum = 0, b = -1, pos = 0;
    for (int bb = 0; bb < B_; ++bb) {
        int ns = num_sent[bb];
        if (i >= cum && i < cum + ns) { b = bb; pos = i - cum; }
        cum += ns;
    }
    if (b >= 0 && pos < S_) t0[((size_t)b * S_ + pos) * F_ + f] = s0[idx];
}

// ---------------------------------------------------------------------------
// Phase F: LSTM over sentence axis.  One block per batch row; thread = unit.
// Weights pre-transposed to [k][j] for coalesced access.
// ---------------------------------------------------------------------------
__global__ __launch_bounds__(256) void k_lstm(const float* __restrict__ t0,
                                              const float* __restrict__ WihT,
                                              const float* __restrict__ WhhT,
                                              const float* __restrict__ bih,
                                              const float* __restrict__ bhh,
                                              float* __restrict__ m0) {
    __shared__ float hs[H_];
    __shared__ float xs[F_];
    int b = blockIdx.x, u = threadIdx.x;
    float cst = 0.0f;
    hs[u] = 0.0f;
    __syncthreads();
    for (int s = 0; s < S_; ++s) {
        xs[u] = t0[((size_t)b * S_ + s) * F_ + u];
        __syncthreads();
        float g[4];
#pragma unroll
        for (int q = 0; q < 4; ++q) {
            int j = q * H_ + u;                      // gate order: i, f, g, o
            float acc = bih[j] + bhh[j];
            for (int k = 0; k < F_; ++k) acc += xs[k] * WihT[k * 1024 + j];
            for (int k = 0; k < H_; ++k) acc += hs[k] * WhhT[k * 1024 + j];
            g[q] = acc;
        }
        cst = sigf(g[1]) * cst + sigf(g[0]) * tanhf(g[2]);
        float hnew = sigf(g[3]) * tanhf(cst);
        m0[((size_t)b * S_ + s) * H_ + u] = hnew;
        __syncthreads();
        hs[u] = hnew;
        __syncthreads();
    }
}

// ---------------------------------------------------------------------------
// Phase G: masked sentence attention + sigmoid output. One block per batch.
// Online softmax over s for each feature column u.
// ---------------------------------------------------------------------------
__global__ __launch_bounds__(256) void k_final(const float* __restrict__ m0,
                                               const float* __restrict__ Wa1,
                                               const float* __restrict__ ba1,
                                               const float* __restrict__ Wo,
                                               const float* __restrict__ bo,
                                               const float* __restrict__ mask,
                                               float* __restrict__ out) {
    __shared__ float xs[H_];
    __shared__ float red[H_];
    int b = blockIdx.x, u = threadIdx.x;
    float mx = -3.4e38f, d = 0.0f, num = 0.0f;
    for (int s = 0; s < S_; ++s) {
        xs[u] = m0[((size_t)b * S_ + s) * H_ + u];
        __syncthreads();
        float acc = ba1[u];
        for (int k = 0; k < H_; ++k) acc += xs[k] * Wa1[k * H_ + u];
        float l = tanhf(acc);
        if (!(mask[b * S_ + s] > 0.0f)) l = -1e9f;
        float nm = fmaxf(mx, l);
        float sc = expf(mx - nm), e = expf(l - nm);
        d = d * sc + e;
        num = num * sc + e * xs[u];
        mx = nm;
        __syncthreads();
    }
    float s1 = num / d;
    red[u] = s1 * Wo[u];
    __syncthreads();
    for (int off = 128; off > 0; off >>= 1) {
        if (u < off) red[u] += red[u + off];
        __syncthreads();
    }
    if (u == 0) out[b] = sigf(red[0] + bo[0]);
}

// ---------------------------------------------------------------------------
// Host launcher
// ---------------------------------------------------------------------------
static inline size_t alignup(size_t x) { return (x + 255) & ~(size_t)255; }

extern "C" void kernel_launch(void* const* d_in, const int* in_sizes, int n_in,
                              void* d_out, int out_size, void* d_ws, size_t ws_size,
                              hipStream_t stream) {
    const int*   input    = (const int*)  d_in[0];
    const int*   num_sent = (const int*)  d_in[1];
    const float* mask     = (const float*)d_in[2];
    const float* Wemb     = (const float*)d_in[3];
    const float* Wconv    = (const float*)d_in[4];
    const float* bconv    = (const float*)d_in[5];
    const float* Wa0      = (const float*)d_in[6];
    const float* ba0      = (const float*)d_in[7];
    const float* Wih      = (const float*)d_in[8];
    const float* Whh      = (const float*)d_in[9];
    const float* bih      = (const float*)d_in[10];
    const float* bhh      = (const float*)d_in[11];
    const float* Wa1      = (const float*)d_in[12];
    const float* ba1      = (const float*)d_in[13];
    const float* Wo       = (const float*)d_in[14];
    const float* bo       = (const float*)d_in[15];
    float*       out      = (float*)d_out;

    // Workspace layout (region 0 reused: embBf in phase A/B, logits in phase C/D)
    char* ws = (char*)d_ws;
    size_t o = 0;
    const size_t o_emb   = o; o = alignup(o + (size_t)TOTAL_ * SSTRIDE_ * 2);  // >= logits size
    const size_t o_btc   = o; o = alignup(o + (size_t)F_ * KPCONV_ * 2);
    const size_t o_bta   = o; o = alignup(o + (size_t)F_ * F_ * 2);
    const size_t o_c0    = o; o = alignup(o + (size_t)TOTAL_ * TP_ * F_ * 2);
    const size_t o_s0    = o; o = alignup(o + (size_t)TOTAL_ * F_ * 4);
    const size_t o_t0    = o; o = alignup(o + (size_t)B_ * S_ * F_ * 4);
    const size_t o_wiht  = o; o = alignup(o + (size_t)1024 * 256 * 4);
    const size_t o_whht  = o; o = alignup(o + (size_t)1024 * 256 * 4);
    const size_t o_m0    = o; o = alignup(o + (size_t)B_ * S_ * H_ * 4);

    bf16*  embBf  = (bf16*) (ws + o_emb);
    bf16*  logits = (bf16*) (ws + o_emb);   // reuse after conv finishes
    bf16*  BtConv = (bf16*) (ws + o_btc);
    bf16*  BtWa0  = (bf16*) (ws + o_bta);
    bf16*  c0     = (bf16*) (ws + o_c0);
    float* s0     = (float*)(ws + o_s0);
    float* t0     = (float*)(ws + o_t0);
    float* WihT   = (float*)(ws + o_wiht);
    float* WhhT   = (float*)(ws + o_whht);
    float* m0     = (float*)(ws + o_m0);

    // Phase A: prep
    {
        size_t n = (size_t)TOTAL_ * SSTRIDE_;
        k_emb<<<dim3((unsigned)((n + 255) / 256)), dim3(256), 0, stream>>>(input, Wemb, embBf);
    }
    k_btconv<<<dim3((F_ * KPCONV_ + 255) / 256), dim3(256), 0, stream>>>(Wconv, BtConv);
    k_btwa0 <<<dim3((F_ * F_ + 255) / 256),      dim3(256), 0, stream>>>(Wa0, BtWa0);
    k_transpose_w<<<dim3(1024), dim3(256), 0, stream>>>(Wih, WihT);
    k_transpose_w<<<dim3(1024), dim3(256), 0, stream>>>(Whh, WhhT);

    // Phase B: conv GEMM (WMMA, LDS-staged B)
    k_conv<<<dim3(TOTAL_), dim3(512), 0, stream>>>(embBf, BtConv, bconv, c0);
    // Phase C: attention-0 GEMM (WMMA) -> logits (reuses emb region)
    k_attn0<<<dim3(TOTAL_), dim3(512), 0, stream>>>(c0, BtWa0, ba0, logits);
    // Phase D: softmax over time + weighted sum
    k_soft0<<<dim3(TOTAL_), dim3(256), 0, stream>>>(logits, c0, s0);
    // Phase E: scatter to padded [B,S,F]
    k_zero_t0<<<dim3((B_ * S_ * F_ + 255) / 256), dim3(256), 0, stream>>>(t0);
    k_scatter<<<dim3((TOTAL_ * F_ + 255) / 256), dim3(256), 0, stream>>>(s0, num_sent, t0);
    // Phase F: LSTM over sentences
    k_lstm<<<dim3(B_), dim3(256), 0, stream>>>(t0, WihT, WhhT, bih, bhh, m0);
    // Phase G: masked attention + output
    k_final<<<dim3(B_), dim3(256), 0, stream>>>(m0, Wa1, ba1, Wo, bo, mask, out);

    (void)in_sizes; (void)n_in; (void)out_size; (void)ws_size;
}
